// SeparatedAttention_22144851378567
// MI455X (gfx1250) — matrix operand order repaired and checked by reading, not compiled, not run on hardware
//
#include <hip/hip_runtime.h>

// ---------------------------------------------------------------------------
// CDNA5 (gfx1250) SeparatedAttention: bf16 WMMA GEMMs (async-LDS double
// buffered) + fused flash attention
// ---------------------------------------------------------------------------

typedef __attribute__((ext_vector_type(16))) __bf16 bf16x16;
typedef __attribute__((ext_vector_type(8)))  float  f32x8;

union FragAB {
  bf16x16 v;
  unsigned int u[8];
};

__device__ __forceinline__ unsigned short f32_to_bf16(float f) {
  unsigned int u = __float_as_uint(f);
  u += 0x7FFFu + ((u >> 16) & 1u);          // round-to-nearest-even
  return (unsigned short)(u >> 16);
}

__device__ __forceinline__ unsigned int pack_bf16x2(float lo, float hi) {
  return (unsigned int)f32_to_bf16(lo) | ((unsigned int)f32_to_bf16(hi) << 16);
}

__device__ __forceinline__ f32x8 fzero() {
  f32x8 z;
#pragma unroll
  for (int i = 0; i < 8; ++i) z[i] = 0.0f;
  return z;
}

// gfx1250 async global->LDS copy (16B per lane), tracked by ASYNCcnt.
// lds_off = byte offset within workgroup LDS (low 32 bits of generic ptr).
__device__ __forceinline__ void async_load_b128(unsigned lds_off, const void* gaddr) {
  asm volatile("global_load_async_to_lds_b128 %0, %1, off"
               :: "v"(lds_off), "v"(gaddr)
               : "memory");
}
__device__ __forceinline__ void wait_asynccnt0() {
  asm volatile("s_wait_asynccnt 0x0" ::: "memory");
}

// ---------------------------------------------------------------------------
// f32 -> bf16 conversion, 4 elements/thread (all sizes are multiples of 4)
// ---------------------------------------------------------------------------
__global__ void cvt_f32_to_bf16(const float* __restrict__ src,
                                unsigned short* __restrict__ dst, int n4) {
  int i = blockIdx.x * blockDim.x + threadIdx.x;
  int stride = gridDim.x * blockDim.x;
  for (; i < n4; i += stride) {
    float4 f = ((const float4*)src)[i];
    uint2 o;
    o.x = pack_bf16x2(f.x, f.y);
    o.y = pack_bf16x2(f.z, f.w);
    ((uint2*)dst)[i] = o;
  }
}

// ---------------------------------------------------------------------------
// bf16 GEMM: C[M,N] = A[M,K] * B[K,N] (+bias). Out f32 (Cf) or bf16 (Cb).
// Block tile 128x128, 256 threads = 8 waves (4x2), wave tile 32x64,
// K step 32 -> v_wmma_f32_16x16x32_bf16.
// A tile staged with global_load_async_to_lds_b128; B tile staged via regs
// (needs [k][n] -> [n][k] transpose). LDS double buffered.
// N, K multiples of 128/32 (true here); M guarded.
// ---------------------------------------------------------------------------
__global__ __launch_bounds__(256) void gemm_bf16_wmma(
    const unsigned short* __restrict__ A, const unsigned short* __restrict__ Bm,
    const float* __restrict__ bias, float* __restrict__ Cf,
    unsigned short* __restrict__ Cb, int M, int N, int K) {
  __shared__ unsigned short sA[2][128 * 36];   // [m][k], stride 36
  __shared__ unsigned short sB[2][128 * 36];   // [n][k], stride 36 (k contiguous)

  const int tid = threadIdx.x;
  const int wave = tid >> 5, lane = tid & 31;
  const int half = lane >> 4, ln = lane & 15;
  const int wm = (wave >> 1) * 32, wn = (wave & 1) * 64;
  const int mBase = blockIdx.y * 128, nBase = blockIdx.x * 128;

  f32x8 acc[2][4];
#pragma unroll
  for (int i = 0; i < 2; ++i)
#pragma unroll
    for (int j = 0; j < 4; ++j) acc[i][j] = fzero();

  const int bkr = tid >> 3, bnseg = (tid & 7) * 16;

  // --- async stage of A tile: 2 x 16B per thread, rows clamped (OOB rows
  //     are never stored from C, so duplicated data there is harmless) ---
  auto stageA = [&](int buf, int k0) {
#pragma unroll
    for (int i = 0; i < 2; ++i) {
      int row = (tid >> 2) + 64 * i;
      int seg = (tid & 3) * 8;               // in shorts (16B chunks)
      int grow = mBase + row;
      if (grow > M - 1) grow = M - 1;
      const void* g = (const void*)(A + (size_t)grow * K + k0 + seg);
      unsigned lds = (unsigned)(uintptr_t)&sA[buf][row * 36 + seg];
      async_load_b128(lds, g);
    }
  };
  auto loadB = [&](int k0, uint4& b0, uint4& b1) {
    const uint4* g = (const uint4*)(Bm + (size_t)(k0 + bkr) * N + nBase + bnseg);
    b0 = g[0];
    b1 = g[1];
  };
  auto storeB = [&](int buf, uint4 b0, uint4 b1) {
    const unsigned short* h0 = (const unsigned short*)&b0;
    const unsigned short* h1 = (const unsigned short*)&b1;
#pragma unroll
    for (int e = 0; e < 8; ++e) {
      sB[buf][(bnseg + e) * 36 + bkr] = h0[e];
      sB[buf][(bnseg + 8 + e) * 36 + bkr] = h1[e];
    }
  };
  auto compute = [&](int buf) {
    FragAB af[2], bf_[4];
#pragma unroll
    for (int i = 0; i < 2; ++i) {
      const unsigned int* s = (const unsigned int*)sA[buf] + (wm + i * 16 + ln) * 18;
#pragma unroll
      for (int e2 = 0; e2 < 8; ++e2) {
        int Kp = (e2 < 4) ? (8 * half + 2 * e2) : (8 + 8 * half + 2 * e2);
        af[i].u[e2] = s[Kp >> 1];
      }
    }
#pragma unroll
    for (int j = 0; j < 4; ++j) {
      const unsigned int* s = (const unsigned int*)sB[buf] + (wn + j * 16 + ln) * 18;
#pragma unroll
      for (int e2 = 0; e2 < 8; ++e2) bf_[j].u[e2] = s[(e2 < 4) ? (4 * half + e2) : (4 + 4 * half + e2)];
    }
#pragma unroll
    for (int i = 0; i < 2; ++i)
#pragma unroll
      for (int j = 0; j < 4; ++j)
        acc[i][j] = __builtin_amdgcn_wmma_f32_16x16x32_bf16(
            false, af[i].v, false, bf_[j].v, (short)0, acc[i][j], false, false);
  };

  const int nIter = K >> 5;
  // prologue: stage iteration 0 into buffer 0
  stageA(0, 0);
  {
    uint4 b0, b1;
    loadB(0, b0, b1);
    storeB(0, b0, b1);
  }
  wait_asynccnt0();
  __syncthreads();

  uint4 nb0, nb1;
  for (int it = 0; it < nIter; ++it) {
    const int cur = it & 1;
    const bool nxt = (it + 1) < nIter;
    if (nxt) {
      stageA(cur ^ 1, (it + 1) << 5);        // async: overlaps with WMMA below
      loadB((it + 1) << 5, nb0, nb1);
    }
    compute(cur);
    if (nxt) {
      storeB(cur ^ 1, nb0, nb1);
      wait_asynccnt0();
    }
    __syncthreads();
  }

  // --- epilogue ---
#pragma unroll
  for (int i = 0; i < 2; ++i) {
#pragma unroll
    for (int v = 0; v < 8; ++v) {
      int row = mBase + wm + i * 16 + v + 8 * half;
      if (row < M) {
#pragma unroll
        for (int j = 0; j < 4; ++j) {
          int col = nBase + wn + j * 16 + ln;
          float x = acc[i][j][v];
          if (bias) x += bias[col];
          if (Cf) Cf[(size_t)row * N + col] = x;
          else Cb[(size_t)row * N + col] = f32_to_bf16(x);
        }
      }
    }
  }
}

// ---------------------------------------------------------------------------
// Fused flash attention (bf16 in, bf16 out). Computes S^T = K*Q^T and
// O^T = V^T * P^T so softmax stats are per-lane scalars (query = lane&15).
// One workgroup = (b,h, 128-query chunk); each wave owns a 16-query strip.
// Keys streamed in 64-wide LDS chunks; DH=144 zero-padded to 160.
// ---------------------------------------------------------------------------
__global__ __launch_bounds__(256) void attn_fused(
    const unsigned short* __restrict__ Q, int ldq, int qoff,
    const unsigned short* __restrict__ Kt, int ldk, int koff,
    const unsigned short* __restrict__ Vt, int ldv, int voff,
    unsigned short* __restrict__ O, int nq) {
  __shared__ unsigned short sK[64 * 160];   // [key][dh], dh padded w/ zeros
  __shared__ unsigned short sV[144 * 64];   // [dh][key] (transposed)

  const int tid = threadIdx.x;
  const int wave = tid >> 5, lane = tid & 31;
  const int half = lane >> 4, ln = lane & 15;
  const int b = blockIdx.y >> 3, h = blockIdx.y & 7;
  const int qbase = blockIdx.x * 128 + wave * 16;
  const bool active = qbase < nq;   // wave-uniform: EXEC all-ones in WMMA region

  const unsigned short* qp = Q + (size_t)(b * nq) * ldq + qoff + h * 144;
  const unsigned short* kp = Kt + (size_t)(b * 256) * ldk + koff + h * 144;
  const unsigned short* vp = Vt + (size_t)(b * 256) * ldv + voff + h * 144;

  // Q^T B-fragments straight from global (pairs contiguous along dh)
  FragAB bq[5];
  if (active) {
    const int qrow = qbase + ln;
    const bool qv = qrow < nq;
    const unsigned short* qr = qp + (size_t)qrow * ldq;
#pragma unroll
    for (int c = 0; c < 5; ++c)
#pragma unroll
      for (int e2 = 0; e2 < 8; ++e2) {
        int dh = c * 32 + ((e2 < 4) ? (8 * half + 2 * e2) : (8 + 8 * half + 2 * e2));
        bq[c].u[e2] = (qv && dh < 144) ? *(const unsigned int*)(qr + dh) : 0u;
      }
  }

  f32x8 acc[9];   // O^T accum: dh = d*16 + v + 8*half, query = ln
#pragma unroll
  for (int d = 0; d < 9; ++d) acc[d] = fzero();
  float mrun = -3.0e38f, lrun = 0.0f;
  const float cs = 0.0833333333f * 1.44269504f;   // SCALE * log2(e)

  for (int kc = 0; kc < 256; kc += 64) {
    // --- cooperative staging of K chunk and V^T chunk ---
    for (int idx = tid; idx < 64 * 160; idx += 256) {
      int r = idx / 160, c2 = idx - r * 160;
      sK[idx] = (c2 < 144) ? kp[(size_t)(kc + r) * ldk + c2] : (unsigned short)0;
    }
    for (int idx = tid; idx < 144 * 64; idx += 256) {
      int d = idx >> 6, key = idx & 63;
      sV[idx] = vp[(size_t)(kc + key) * ldv + d];
    }
    __syncthreads();

    if (active) {
#pragma unroll
      for (int kb = 0; kb < 64; kb += 32) {
        // ---- S^T = K * Q^T over padded dh (5 x k32) ----
        f32x8 s0 = fzero(), s1 = fzero();
#pragma unroll
        for (int c = 0; c < 5; ++c) {
          FragAB a0, a1;
          const unsigned int* k0r = (const unsigned int*)sK + (kb + ln) * 80;
          const unsigned int* k1r = (const unsigned int*)sK + (kb + 16 + ln) * 80;
#pragma unroll
          for (int e2 = 0; e2 < 8; ++e2) {
            int Kp = c * 32 + ((e2 < 4) ? (8 * half + 2 * e2) : (8 + 8 * half + 2 * e2));
            a0.u[e2] = k0r[Kp >> 1];
            a1.u[e2] = k1r[Kp >> 1];
          }
          s0 = __builtin_amdgcn_wmma_f32_16x16x32_bf16(false, a0.v, false, bq[c].v,
                                                       (short)0, s0, false, false);
          s1 = __builtin_amdgcn_wmma_f32_16x16x32_bf16(false, a1.v, false, bq[c].v,
                                                       (short)0, s1, false, false);
        }
        // ---- online softmax (per-lane: query = ln) ----
        float bmax = -3.0e38f;
#pragma unroll
        for (int i = 0; i < 8; ++i) bmax = fmaxf(bmax, fmaxf(s0[i], s1[i]));
        bmax = fmaxf(bmax, __shfl_xor(bmax, 16, 32));
        const float mnew = fmaxf(mrun, bmax);
        const float alpha = exp2f((mrun - mnew) * cs);
#pragma unroll
        for (int d = 0; d < 9; ++d)
#pragma unroll
          for (int i = 0; i < 8; ++i) acc[d][i] *= alpha;
        float p0[8], p1[8], bsum = 0.0f;
#pragma unroll
        for (int i = 0; i < 8; ++i) {
          p0[i] = exp2f((s0[i] - mnew) * cs);
          p1[i] = exp2f((s1[i] - mnew) * cs);
          bsum += p0[i] + p1[i];
        }
        bsum += __shfl_xor(bsum, 16, 32);
        lrun = lrun * alpha + bsum;
        mrun = mnew;

        // ---- assemble P^T B-frag (32 keys x 16 queries) via half-swap ----
        float x0[8], x1[8];
#pragma unroll
        for (int i = 0; i < 8; ++i) {
          x0[i] = __shfl_xor(p0[i], 16, 32);
          x1[i] = __shfl_xor(p1[i], 16, 32);
        }
        FragAB pb;
#pragma unroll
        for (int j = 0; j < 4; ++j) {
          float e0 = half ? x1[2 * j] : p0[2 * j];
          float e1 = half ? x1[2 * j + 1] : p0[2 * j + 1];
          float e8 = half ? p1[2 * j] : x0[2 * j];
          float e9 = half ? p1[2 * j + 1] : x0[2 * j + 1];
          pb.u[j] = pack_bf16x2(e0, e1);
          pb.u[4 + j] = pack_bf16x2(e8, e9);
        }
        // ---- O^T += V^T * P^T (9 dh strips) ----
#pragma unroll
        for (int d = 0; d < 9; ++d) {
          FragAB av;
          const unsigned int* vr = (const unsigned int*)sV + (d * 16 + ln) * 32;
#pragma unroll
          for (int e2 = 0; e2 < 8; ++e2) {
            int Kp = kb + 16 * half + 2 * e2;
            av.u[e2] = vr[Kp >> 1];
          }
          acc[d] = __builtin_amdgcn_wmma_f32_16x16x32_bf16(
              false, av.v, false, pb.v, (short)0, acc[d], false, false);
        }
      }
    }
    __syncthreads();
  }

  if (active) {
    const int qrow = qbase + ln;
    if (qrow < nq) {
      const float inv = 1.0f / lrun;
      unsigned short* orow = O + (size_t)(b * nq + qrow) * 1152 + h * 144;
#pragma unroll
      for (int d = 0; d < 9; ++d)
#pragma unroll
        for (int i = 0; i < 8; ++i)
          orow[d * 16 + i + 8 * half] = f32_to_bf16(acc[d][i] * inv);
    }
  }
}

// ---------------------------------------------------------------------------
// Host-side orchestration
// ---------------------------------------------------------------------------
extern "C" void kernel_launch(void* const* d_in, const int* in_sizes, int n_in,
                              void* d_out, int out_size, void* d_ws, size_t ws_size,
                              hipStream_t stream) {
  (void)in_sizes; (void)n_in; (void)out_size; (void)ws_size;
  const float* act = (const float*)d_in[0];
  const float* non = (const float*)d_in[1];
  const float* Wqkv = (const float*)d_in[2];
  const float* Wq = (const float*)d_in[3];
  const float* Wkv = (const float*)d_in[4];
  const float* Wpn = (const float*)d_in[5];
  const float* bpn = (const float*)d_in[6];
  const float* Wpa = (const float*)d_in[7];
  const float* bpa = (const float*)d_in[8];

  float* out_act = (float*)d_out;                        // [3648,1152]
  float* out_non = out_act + (size_t)3648 * 1152;        // [16384,1152]

  char* w = (char*)d_ws;
  auto alloc = [&](size_t elems) {
    unsigned short* p = (unsigned short*)w;
    w += ((elems * 2 + 255) & ~(size_t)255);
    return p;
  };
  unsigned short* bfNon = alloc((size_t)16384 * 1152);
  unsigned short* bfAct = alloc((size_t)3648 * 1152);
  unsigned short* bfWqkv = alloc((size_t)1152 * 3456);
  unsigned short* bfWq = alloc((size_t)1152 * 1152);
  unsigned short* bfWkv = alloc((size_t)1152 * 2304);
  unsigned short* bfWpn = alloc((size_t)1152 * 1152);
  unsigned short* bfWpa = alloc((size_t)1152 * 1152);
  unsigned short* qkv = alloc((size_t)16384 * 3456);
  unsigned short* qact = alloc((size_t)3648 * 1152);
  unsigned short* kvact = alloc((size_t)16384 * 2304);
  unsigned short* aoN = alloc((size_t)16384 * 1152);
  unsigned short* aoA = alloc((size_t)3648 * 1152);

  auto cvt = [&](const float* s, unsigned short* d, int n) {
    int n4 = n / 4;
    int blocks = (n4 + 255) / 256;
    if (blocks > 4096) blocks = 4096;
    hipLaunchKernelGGL(cvt_f32_to_bf16, dim3(blocks), dim3(256), 0, stream, s, d, n4);
  };
  cvt(non, bfNon, 16384 * 1152);
  cvt(act, bfAct, 3648 * 1152);
  cvt(Wqkv, bfWqkv, 1152 * 3456);
  cvt(Wq, bfWq, 1152 * 1152);
  cvt(Wkv, bfWkv, 1152 * 2304);
  cvt(Wpn, bfWpn, 1152 * 1152);
  cvt(Wpa, bfWpa, 1152 * 1152);

  auto gemm = [&](const unsigned short* Am, const unsigned short* Bm,
                  const float* bias, float* Cf, unsigned short* Cb,
                  int M, int N, int K) {
    dim3 grid(N / 128, (M + 127) / 128);
    hipLaunchKernelGGL(gemm_bf16_wmma, grid, dim3(256), 0, stream,
                       Am, Bm, bias, Cf, Cb, M, N, K);
  };

  // projections into intermediates
  gemm(bfNon, bfWqkv, nullptr, nullptr, qkv, 16384, 3456, 1152);   // qkv_non
  gemm(bfAct, bfWq, nullptr, nullptr, qact, 3648, 1152, 1152);     // q_act
  gemm(bfNon, bfWkv, nullptr, nullptr, kvact, 16384, 2304, 1152);  // kv_act

  // attention: non-act self-attn (Q at col 0, K at 1152, V at 2304 within qkv row)
  hipLaunchKernelGGL(attn_fused, dim3(2, 64 * 8), dim3(256), 0, stream,
                     qkv, 3456, 0, qkv, 3456, 1152, qkv, 3456, 2304, aoN, 256);
  // attention: act cross-attn (K at col 0, V at 1152 within kv row)
  hipLaunchKernelGGL(attn_fused, dim3(1, 64 * 8), dim3(256), 0, stream,
                     qact, 1152, 0, kvact, 2304, 0, kvact, 2304, 1152, aoA, 57);

  // output projections (+bias), f32 outputs
  gemm(aoN, bfWpn, bpn, out_non, nullptr, 16384, 1152, 1152);
  gemm(aoA, bfWpa, bpa, out_act, nullptr, 3648, 1152, 1152);
}
